// MultiHeadAttention_33578054320717
// MI455X (gfx1250) — compile-verified, matrix operand order checked by reading, not run on hardware
//
#include <hip/hip_runtime.h>

typedef __attribute__((ext_vector_type(16))) _Float16 v16h;
typedef __attribute__((ext_vector_type(8)))  float    v8f;

#define WMMA(a, b, c) __builtin_amdgcn_wmma_f32_16x16x32_f16( \
    false, (a), false, (b), (short)0, (c), false, false)

// Problem constants
static constexpr int CB = 4;     // batch
static constexpr int CK = 2048;  // time window
static constexpr int CD = 512;   // d_model
static constexpr int CH = 8;     // heads
// per-head q/v dim = 64

// ---------------------------------------------------------------------------
// Kernel 0: weight transpose + f32->f16 convert.
//   W_q/W_k/W_v: [H, D, 64] -> w*t: [H, 64, D]   (Bt row-major)
//   W_o:         [512, 512] -> wot: [512 outcol][512 kin]
// ---------------------------------------------------------------------------
__global__ void wprep(const float* __restrict__ Wq, const float* __restrict__ Wk,
                      const float* __restrict__ Wv, const float* __restrict__ Wo,
                      _Float16* __restrict__ wqt, _Float16* __restrict__ wkt,
                      _Float16* __restrict__ wvt, _Float16* __restrict__ wot) {
  int i = blockIdx.x * 256 + threadIdx.x;
  const int per = CH * CD * 64;  // 262144
  if (i < per) {
    int h = i / (64 * CD);
    int rem = i % (64 * CD);
    int c = rem / CD;       // out col within head
    int d = rem % CD;       // input dim
    size_t src = (size_t)(h * CD + d) * 64 + c;
    wqt[i] = (_Float16)Wq[src];
    wkt[i] = (_Float16)Wk[src];
    wvt[i] = (_Float16)Wv[src];
  } else {
    int j = i - per;        // 0..262143
    int c = j / CD;         // out col
    int d = j % CD;         // in dim
    wot[j] = (_Float16)Wo[(size_t)d * CD + c];
  }
}

// Convert 16 consecutive f32 to a v16h A/B operand fragment
__device__ inline v16h cvtA(const float* __restrict__ p) {
  const float4* p4 = (const float4*)p;
  float4 f0 = p4[0], f1 = p4[1], f2 = p4[2], f3 = p4[3];
  v16h a;
  a[0]  = (_Float16)f0.x; a[1]  = (_Float16)f0.y; a[2]  = (_Float16)f0.z; a[3]  = (_Float16)f0.w;
  a[4]  = (_Float16)f1.x; a[5]  = (_Float16)f1.y; a[6]  = (_Float16)f1.z; a[7]  = (_Float16)f1.w;
  a[8]  = (_Float16)f2.x; a[9]  = (_Float16)f2.y; a[10] = (_Float16)f2.z; a[11] = (_Float16)f2.w;
  a[12] = (_Float16)f3.x; a[13] = (_Float16)f3.y; a[14] = (_Float16)f3.z; a[15] = (_Float16)f3.w;
  return a;
}

// ---------------------------------------------------------------------------
// Kernel 1: QKV projection, one matrix per grid.z (0=q, 1=k, 2=v).
// grid = (B*K/64 row tiles, H, 3), block = 32 (one wave).
// Each wave: 64 rows x 64 cols, K-dim = 512. 16 accum tiles; each B fragment
// is reused by 4 WMMAs (4 row sub-tiles).
//   qp[b,h][row][64], kp[b,h][row][64]  (row-major, f16)
//   vt[b,h][vcol][2048]                 (V transposed, f16)
// ---------------------------------------------------------------------------
__global__ void __launch_bounds__(32) proj(
    const float* __restrict__ Xq, const float* __restrict__ Xk, const float* __restrict__ Xv,
    const _Float16* __restrict__ wqt, const _Float16* __restrict__ wkt, const _Float16* __restrict__ wvt,
    const float* __restrict__ bq, const float* __restrict__ bk, const float* __restrict__ bv,
    _Float16* __restrict__ qp, _Float16* __restrict__ kp, _Float16* __restrict__ vt) {
  const int rt = blockIdx.x;         // 64-row tile over B*K (0..127)
  const int h  = blockIdx.y;
  const int z  = blockIdx.z;         // 0=q, 1=k, 2=v
  const int lane = threadIdx.x & 31;
  const int lr = lane & 15, lh = lane >> 4;
  const int row0 = rt * 64;          // global row in [0, 8192)
  const int b = row0 / CK;
  const int krow0 = row0 % CK;

  const float*    X    = (z == 0) ? Xq  : (z == 1) ? Xk  : Xv;
  const _Float16* wt   = (z == 0) ? wqt : (z == 1) ? wkt : wvt;
  const float*    bias = (z == 0) ? bq  : (z == 1) ? bk  : bv;

  v8f acc[4][4] = {};  // [row sub-tile][col tile]

  const _Float16* wh = wt + (size_t)h * 64 * CD;

  for (int s = 0; s < 16; ++s) {
    const int k0 = s * 32 + lh * 16;
    v16h a[4];
#pragma unroll
    for (int r2 = 0; r2 < 4; ++r2)
      a[r2] = cvtA(X + (size_t)(row0 + 16 * r2 + lr) * CD + k0);
#pragma unroll
    for (int t = 0; t < 4; ++t) {
      v16h bt = *(const v16h*)(wh + (size_t)(t * 16 + lr) * CD + k0);
#pragma unroll
      for (int r2 = 0; r2 < 4; ++r2)
        acc[r2][t] = WMMA(a[r2], bt, acc[r2][t]);
    }
  }

  const size_t bh = (size_t)(b * CH + h);
  if (z < 2) {
    _Float16* dst = ((z == 0) ? qp : kp) + bh * CK * 64;
#pragma unroll
    for (int t = 0; t < 4; ++t) {
      const int c = t * 16 + lr;
      float bs = bias[h * 64 + c];
#pragma unroll
      for (int r2 = 0; r2 < 4; ++r2)
#pragma unroll
        for (int r = 0; r < 8; ++r) {
          int m = krow0 + 16 * r2 + r + 8 * lh;
          dst[(size_t)m * 64 + c] = (_Float16)(acc[r2][t][r] + bs);
        }
    }
  } else {
    _Float16* dst = vt + bh * 64 * CK;   // transposed: [vcol][kpos]
#pragma unroll
    for (int t = 0; t < 4; ++t) {
      const int c = t * 16 + lr;
      float bs = bias[h * 64 + c];
#pragma unroll
      for (int r2 = 0; r2 < 4; ++r2)
#pragma unroll
        for (int r = 0; r < 8; ++r) {
          int m = krow0 + 16 * r2 + r + 8 * lh;
          dst[(size_t)c * CK + m] = (_Float16)(acc[r2][t][r] + bs);
        }
    }
  }
}

// ---------------------------------------------------------------------------
// Kernel 2: causal flash attention, one wave per 16-row query tile per (b,h).
// grid = (128 qtiles, 8 heads, 4 batch), block = 32.
// Output heads buffer ho: [B, K, H*64] f16.
// ---------------------------------------------------------------------------
__global__ void __launch_bounds__(32) attn(
    const _Float16* __restrict__ qp, const _Float16* __restrict__ kp,
    const _Float16* __restrict__ vt, _Float16* __restrict__ ho) {
  __shared__ __align__(32) _Float16 lds[16 * 32];
  const int qt = blockIdx.x;
  const int h  = blockIdx.y;
  const int bz = blockIdx.z;
  const int lane = threadIdx.x & 31;
  const int lr = lane & 15, lh = lane >> 4;
  const int q0 = qt * 16;
  const size_t bh = (size_t)(bz * CH + h);
  const _Float16* qbase = qp + bh * CK * 64;
  const _Float16* kbase = kp + bh * CK * 64;
  const _Float16* vbase = vt + bh * 64 * CK;

  // Q tile A-operands (K-dim 0..31 and 32..63)
  const v16h qa0 = *(const v16h*)(qbase + (size_t)(q0 + lr) * 64 + lh * 16);
  const v16h qa1 = *(const v16h*)(qbase + (size_t)(q0 + lr) * 64 + 32 + lh * 16);

  v8f o[4] = {};
  float m_[8], l_[8];
#pragma unroll
  for (int r = 0; r < 8; ++r) { m_[r] = -3.0e38f; l_[r] = 0.0f; }

  const float scale = 0.022097086912079608f;  // 1/sqrt(2048)
  const float NEG = -3.0e38f;

  for (int n0 = 0; n0 < q0 + 16; n0 += 32) {
    // Prefetch next key/value block into cache while we compute this one.
    if (n0 + 32 < q0 + 16) {
      __builtin_prefetch(kbase + (size_t)(n0 + 32 + lr) * 64 + lh * 16, 0, 3);
      __builtin_prefetch(kbase + (size_t)(n0 + 48 + lr) * 64 + lh * 16, 0, 3);
      __builtin_prefetch(vbase + (size_t)(lr + 16 * lh) * CK + n0 + 32, 0, 3);
    }
    // S = Q * K^T for a 16x32 key block (two 16-col tiles)
    v8f s0 = {}, s1 = {};
    {
      v16h kb;
      kb = *(const v16h*)(kbase + (size_t)(n0 + lr) * 64 + lh * 16);
      s0 = WMMA(qa0, kb, s0);
      kb = *(const v16h*)(kbase + (size_t)(n0 + lr) * 64 + 32 + lh * 16);
      s0 = WMMA(qa1, kb, s0);
      kb = *(const v16h*)(kbase + (size_t)(n0 + 16 + lr) * 64 + lh * 16);
      s1 = WMMA(qa0, kb, s1);
      kb = *(const v16h*)(kbase + (size_t)(n0 + 16 + lr) * 64 + 32 + lh * 16);
      s1 = WMMA(qa1, kb, s1);
    }
    const bool needmask = (n0 + 32 > q0);
    float corr[8];
#pragma unroll
    for (int r = 0; r < 8; ++r) {
      float a  = s0[r] * scale;
      float b2 = s1[r] * scale;
      if (needmask) {
        int mg = q0 + r + 8 * lh;
        if (n0 + lr      > mg) a  = NEG;
        if (n0 + 16 + lr > mg) b2 = NEG;
      }
      // row max across the 16 lanes sharing this row
      float t = fmaxf(a, b2);
      t = fmaxf(t, __shfl_xor(t, 1));
      t = fmaxf(t, __shfl_xor(t, 2));
      t = fmaxf(t, __shfl_xor(t, 4));
      t = fmaxf(t, __shfl_xor(t, 8));
      float mnew = fmaxf(m_[r], t);
      corr[r] = __expf(m_[r] - mnew);
      m_[r] = mnew;
      a  = __expf(a - mnew);
      b2 = __expf(b2 - mnew);
      s0[r] = a;
      s1[r] = b2;
      float rs = a + b2;
      rs += __shfl_xor(rs, 1);
      rs += __shfl_xor(rs, 2);
      rs += __shfl_xor(rs, 4);
      rs += __shfl_xor(rs, 8);
      l_[r] = l_[r] * corr[r] + rs;
    }
    // rescale running output
#pragma unroll
    for (int t = 0; t < 4; ++t)
#pragma unroll
      for (int r = 0; r < 8; ++r) o[t][r] *= corr[r];

    // C-layout f32 probs -> LDS -> A-layout f16
    __syncthreads();
#pragma unroll
    for (int r = 0; r < 8; ++r) {
      lds[(r + 8 * lh) * 32 + lr]      = (_Float16)s0[r];
      lds[(r + 8 * lh) * 32 + 16 + lr] = (_Float16)s1[r];
    }
    __syncthreads();
    const v16h pa = *(const v16h*)(lds + lr * 32 + lh * 16);
#pragma unroll
    for (int t = 0; t < 4; ++t) {
      v16h vb = *(const v16h*)(vbase + (size_t)(t * 16 + lr) * CK + n0 + lh * 16);
      o[t] = WMMA(pa, vb, o[t]);
    }
  }

  // normalize and write heads buffer [B, K, H*64]
  _Float16* hob = ho + ((size_t)(bz * CK + q0)) * (CH * 64) + h * 64;
#pragma unroll
  for (int r = 0; r < 8; ++r) {
    float inv = 1.0f / l_[r];
#pragma unroll
    for (int t = 0; t < 4; ++t)
      hob[(size_t)(r + 8 * lh) * (CH * 64) + t * 16 + lr] = (_Float16)(o[t][r] * inv);
  }
}

// ---------------------------------------------------------------------------
// Kernel 3: output projection [8192,512] x [512,512] + bias -> f32 out.
// grid = (128 row tiles of 64, 8 col tiles of 64), block = 32.
// 16 accum tiles; every A and B fragment is reused by 4 WMMAs.
// ---------------------------------------------------------------------------
__global__ void __launch_bounds__(32) outproj(
    const _Float16* __restrict__ ho, const _Float16* __restrict__ wot,
    const float* __restrict__ bo, float* __restrict__ out) {
  const int rt = blockIdx.x;
  const int ct = blockIdx.y;
  const int lane = threadIdx.x & 31;
  const int lr = lane & 15, lh = lane >> 4;
  const int row0 = rt * 64, c0 = ct * 64;

  v8f acc[4][4] = {};
  for (int s = 0; s < 16; ++s) {
    const int k0 = s * 32 + lh * 16;
    v16h a[4];
#pragma unroll
    for (int r2 = 0; r2 < 4; ++r2)
      a[r2] = *(const v16h*)(ho + (size_t)(row0 + 16 * r2 + lr) * CD + k0);
#pragma unroll
    for (int t = 0; t < 4; ++t) {
      v16h bv = *(const v16h*)(wot + (size_t)(c0 + t * 16 + lr) * CD + k0);
#pragma unroll
      for (int r2 = 0; r2 < 4; ++r2)
        acc[r2][t] = WMMA(a[r2], bv, acc[r2][t]);
    }
  }
#pragma unroll
  for (int t = 0; t < 4; ++t) {
    float bs = bo[c0 + t * 16 + lr];
#pragma unroll
    for (int r2 = 0; r2 < 4; ++r2)
#pragma unroll
      for (int r = 0; r < 8; ++r)
        out[(size_t)(row0 + 16 * r2 + r + 8 * lh) * CD + c0 + t * 16 + lr] =
            acc[r2][t][r] + bs;
  }
}

// ---------------------------------------------------------------------------
extern "C" void kernel_launch(void* const* d_in, const int* in_sizes, int n_in,
                              void* d_out, int out_size, void* d_ws, size_t ws_size,
                              hipStream_t stream) {
  const float* query = (const float*)d_in[0];
  const float* key_  = (const float*)d_in[1];
  const float* value = (const float*)d_in[2];
  const float* W_q   = (const float*)d_in[3];
  const float* b_q   = (const float*)d_in[4];
  const float* W_k   = (const float*)d_in[5];
  const float* b_k   = (const float*)d_in[6];
  const float* W_v   = (const float*)d_in[7];
  const float* b_v   = (const float*)d_in[8];
  const float* W_o   = (const float*)d_in[9];
  const float* b_o   = (const float*)d_in[10];
  float* out = (float*)d_out;

  char* w = (char*)d_ws;
  const size_t MB = 1u << 20;
  _Float16* qp  = (_Float16*)(w);                            // 8 MB
  _Float16* kp  = (_Float16*)(w + 8 * MB);                   // 8 MB
  _Float16* vt  = (_Float16*)(w + 16 * MB);                  // 8 MB
  _Float16* wqt = (_Float16*)(w + 24 * MB);                  // 512 KB
  _Float16* wkt = (_Float16*)(w + 24 * MB + 1 * 524288);
  _Float16* wvt = (_Float16*)(w + 24 * MB + 2 * 524288);
  _Float16* wot = (_Float16*)(w + 24 * MB + 3 * 524288);
  _Float16* ho  = (_Float16*)(w + 26 * MB);                  // 8 MB

  wprep<<<2048, 256, 0, stream>>>(W_q, W_k, W_v, W_o, wqt, wkt, wvt, wot);
  proj<<<dim3(CB * CK / 64, CH, 3), 32, 0, stream>>>(query, key_, value,
                                                     wqt, wkt, wvt, b_q, b_k, b_v,
                                                     qp, kp, vt);
  attn<<<dim3(CK / 16, CH, CB), 32, 0, stream>>>(qp, kp, vt, ho);
  outproj<<<dim3(CB * CK / 64, CD / 64), 32, 0, stream>>>(ho, wot, b_o, out);
}
